// Attention_24781961298609
// MI455X (gfx1250) — matrix operand order repaired and checked by reading, not compiled, not run on hardware
//
#include <hip/hip_runtime.h>

// ---------------------------------------------------------------------------
// Types for CDNA5 WMMA
// ---------------------------------------------------------------------------
typedef __bf16 v16bf __attribute__((ext_vector_type(16)));
typedef float  v8f   __attribute__((ext_vector_type(8)));
typedef unsigned int v4u __attribute__((ext_vector_type(4)));

struct frag_bits { v4u a, b; };

// Problem constants
#define BB   2
#define SS   2048
#define DD   2048
#define HH   32
#define KVH  8
#define HDH  64

// ---------------------------------------------------------------------------
// bf16 helpers (bit-level, round-to-nearest-even)
// ---------------------------------------------------------------------------
__device__ inline unsigned short f2bf(float f) {
    unsigned u = __builtin_bit_cast(unsigned, f);
    u += 0x7fffu + ((u >> 16) & 1u);
    return (unsigned short)(u >> 16);
}
__device__ inline float bf2f(unsigned short h) {
    return __builtin_bit_cast(float, (unsigned)h << 16);
}

// Load a 16-element bf16 fragment from two 16-byte-aligned 8-element chunks.
__device__ inline v16bf make_frag(const unsigned short* p0, const unsigned short* p1) {
    frag_bits s;
    s.a = *(const v4u*)p0;
    s.b = *(const v4u*)p1;
    return __builtin_bit_cast(v16bf, s);
}

__device__ inline v8f wmma_bf16(v16bf a, v16bf b, v8f c) {
    return __builtin_amdgcn_wmma_f32_16x16x32_bf16(false, a, false, b, (short)0, c,
                                                   false, false);
}

// CDNA5 async global->LDS copy (ASYNCcnt-tracked DMA, 16B per lane).
__device__ inline void async_b128(unsigned lds_off, unsigned long long gaddr) {
    asm volatile("global_load_async_to_lds_b128 %0, %1, off"
                 :: "v"(lds_off), "v"(gaddr)
                 : "memory");
}
__device__ inline void wait_async0() {
    asm volatile("s_wait_asynccnt 0x0" ::: "memory");
}

// 16-lane row reductions (C/D layout: cols live in lanes 0..15 of each half)
__device__ inline float redmax16(float v) {
    v = fmaxf(v, __shfl_xor(v, 1, 32));
    v = fmaxf(v, __shfl_xor(v, 2, 32));
    v = fmaxf(v, __shfl_xor(v, 4, 32));
    v = fmaxf(v, __shfl_xor(v, 8, 32));
    return v;
}
__device__ inline float redsum16(float v) {
    v += __shfl_xor(v, 1, 32);
    v += __shfl_xor(v, 2, 32);
    v += __shfl_xor(v, 4, 32);
    v += __shfl_xor(v, 8, 32);
    return v;
}

// ---------------------------------------------------------------------------
// fp32 -> bf16 streaming conversion, 8 elements per thread.
// ---------------------------------------------------------------------------
__global__ __launch_bounds__(256) void cvt_bf16_kernel(
    const float* __restrict__ src, unsigned short* __restrict__ dst, int n8) {
    const int i = blockIdx.x * blockDim.x + threadIdx.x;
    if (i >= n8) return;
    const int base = i * 8;
#pragma unroll
    for (int j = 0; j < 8; ++j) dst[base + j] = f2bf(src[base + j]);
}

// ---------------------------------------------------------------------------
// GEMM: C(M,N) = A(M,K) @ W(N,K)^T, bf16 WMMA, fp32 accumulate.
// Block tile 128x64, 8 waves, each wave a 32x32 region (4 WMMA / 32-k-step).
// A and W are bf16. Tiles staged by async global->LDS DMA, double-buffered:
// next tile's DMA is in flight while the current tile is consumed by WMMA.
// M%128==0, N%64==0, K%32==0 (guaranteed here).
// ---------------------------------------------------------------------------
template <typename CT>
__global__ __launch_bounds__(256) void gemm_bf16_kernel(
    const unsigned short* __restrict__ A, const unsigned short* __restrict__ W,
    CT* __restrict__ C, int M, int N, int K) {
    // row stride 40 ushorts (80B): 64B data + 16B pad, keeps 16B alignment
    __shared__ unsigned short As[2][128][40];
    __shared__ unsigned short Ws[2][64][40];

    const int tid  = threadIdx.x;
    const int lane = tid & 31;
    const int wave = tid >> 5;
    const int bm = blockIdx.y, bn = blockIdx.x;
    const int waveM = wave >> 1;   // 0..3
    const int waveN = wave & 1;    // 0..1

    const unsigned asb = (unsigned)(unsigned long long)&As[0][0][0];
    const unsigned wsb = (unsigned)(unsigned long long)&Ws[0][0][0];

    const unsigned short* Abase = A + (size_t)(bm * 128) * K;
    const unsigned short* Wbase = W + (size_t)(bn * 64) * K;

    auto issue_tile = [&](int buf, int k0) {
        // A tile: 128 rows x 32 bf16 = 512 chunks of 16B
#pragma unroll
        for (int c = tid; c < 512; c += 256) {
            const int row = c >> 2, seg = c & 3;
            async_b128(asb + (unsigned)buf * 10240u + row * 80u + seg * 16u,
                       (unsigned long long)(Abase + (size_t)row * K + k0 + seg * 8));
        }
        // W tile: 64 rows x 32 bf16 = 256 chunks of 16B
        {
            const int row = tid >> 2, seg = tid & 3;
            async_b128(wsb + (unsigned)buf * 5120u + row * 80u + seg * 16u,
                       (unsigned long long)(Wbase + (size_t)row * K + k0 + seg * 8));
        }
    };

    v8f acc[2][2] = {};

    const int qb  = (lane >> 4) * 8;   // A-fragment K sub-offset
    const int kb2 = (lane >> 4) * 16;  // unused here; B fragment also uses qb
    const int r15 = lane & 15;
    const int nk  = K / 32;

    issue_tile(0, 0);
    for (int ks = 0; ks < nk; ++ks) {
        wait_async0();        // this wave's DMA for tile `ks` has landed
        __syncthreads();      // everyone's DMA has landed
        if (ks + 1 < nk) issue_tile((ks + 1) & 1, (ks + 1) * 32);  // overlap
        const int buf = ks & 1;

        v16bf aF[2], bF[2];
#pragma unroll
        for (int ti = 0; ti < 2; ++ti) {
            const unsigned short* p = &As[buf][waveM * 32 + ti * 16 + r15][0];
            aF[ti] = make_frag(p + qb, p + 16 + qb);
        }
#pragma unroll
        for (int tj = 0; tj < 2; ++tj) {
            const unsigned short* p = &Ws[buf][waveN * 32 + tj * 16 + r15][0];
            bF[tj] = make_frag(p + qb, p + 16 + qb);
        }
#pragma unroll
        for (int ti = 0; ti < 2; ++ti)
#pragma unroll
            for (int tj = 0; tj < 2; ++tj)
                acc[ti][tj] = wmma_bf16(aF[ti], bF[tj], acc[ti][tj]);

        __syncthreads();      // done reading buf before it is refilled
    }

    // Epilogue: D layout -> global
#pragma unroll
    for (int ti = 0; ti < 2; ++ti)
#pragma unroll
        for (int tj = 0; tj < 2; ++tj)
#pragma unroll
            for (int j = 0; j < 8; ++j) {
                const int row = bm * 128 + waveM * 32 + ti * 16 + (lane >> 4) * 8 + j;
                const int col = bn * 64 + waveN * 32 + tj * 16 + r15;
                const float v = acc[ti][tj][j];
                if constexpr (__is_same(CT, float))
                    C[(size_t)row * N + col] = v;
                else
                    C[(size_t)row * N + col] = f2bf(v);
            }
}

// ---------------------------------------------------------------------------
// RoPE in place on qtmp (B*S, H*HD) and ktmp (B*S, KV*HD), bf16 storage.
// ---------------------------------------------------------------------------
__global__ __launch_bounds__(256) void rope_kernel(
    unsigned short* __restrict__ q, unsigned short* __restrict__ k,
    const float* __restrict__ fc, const float* __restrict__ fs) {
    const int total = BB * SS * (HH + KVH) * (HDH / 2);
    int i = blockIdx.x * blockDim.x + threadIdx.x;
    if (i >= total) return;
    const int p   = i & 31;           // pair index within head dim (HD/2 = 32)
    const int r   = i >> 5;
    const int hh  = r % (HH + KVH);   // 0..39 : 0-31 => q head, 32-39 => k head
    const int pos = r / (HH + KVH);   // 0..B*S-1
    const int s   = pos & (SS - 1);

    unsigned short* base;
    if (hh < HH) base = q + (size_t)pos * (HH * HDH) + hh * HDH + p * 2;
    else         base = k + (size_t)pos * (KVH * HDH) + (hh - HH) * HDH + p * 2;

    const float tr = bf2f(base[0]);
    const float ti = bf2f(base[1]);
    const float c  = fc[s * 32 + p];
    const float sn = fs[s * 32 + p];
    base[0] = f2bf(tr * c - ti * sn);
    base[1] = f2bf(tr * sn + ti * c);
}

// ---------------------------------------------------------------------------
// Causal GQA flash attention. One wave handles a 16-query tile for one head.
// Q: qtmp (b*S+s, H*HD) bf16 (post-RoPE). K: ktmp (b*S+s, KV*HD). V: vtmp same.
// Output: oattn (b*S+s, H*HD) bf16 = softmax(QK^T/8, causal) @ V.
// ---------------------------------------------------------------------------
__global__ __launch_bounds__(256) void attn_kernel(
    const unsigned short* __restrict__ qv, const unsigned short* __restrict__ kv,
    const unsigned short* __restrict__ vv, unsigned short* __restrict__ ov) {
    __shared__ unsigned short Plds[8][16][40];  // per-wave P relayout buffer

    const int lane = threadIdx.x & 31;
    const int wave = threadIdx.x >> 5;
    const int widx = blockIdx.x * 8 + wave;     // 0 .. B*H*(S/16)-1
    const int qblk = widx & 127;                // S/16 = 128
    const int h    = (widx >> 7) & 31;
    const int b    = widx >> 12;
    const int g    = h >> 2;                    // kv head (REP = 4)
    const float scale = 0.125f;                 // 1/sqrt(64)

    const int r15 = lane & 15;
    const int qb  = (lane >> 4) * 8;
    const int kb2 = (lane >> 4) * 16;

    // Q fragments (A layout): row = lane&15, K-dim = head dim (64 -> 2 frags)
    const unsigned short* qr =
        qv + (size_t)(b * SS + qblk * 16 + r15) * (HH * HDH) + h * HDH;
    const v16bf q0 = make_frag(qr + qb,      qr + 16 + qb);
    const v16bf q1 = make_frag(qr + 32 + qb, qr + 48 + qb);

    v8f oacc[4] = {};
    float mrow[8], lrow[8];
#pragma unroll
    for (int j = 0; j < 8; ++j) { mrow[j] = -1e30f; lrow[j] = 0.0f; }

    const int qrow0 = qblk * 16 + (lane >> 4) * 8;  // + j gives this lane's rows
    const int kend  = qblk * 16 + 16;

    for (int kb32 = 0; kb32 < kend; kb32 += 32) {
        // ---- scores: two 16x16 tiles over 32 keys, K-reduce over HD=64 ----
        v8f st[2] = {};
#pragma unroll
        for (int t = 0; t < 2; ++t) {
            const int key = kb32 + t * 16 + r15;
            const unsigned short* kr =
                kv + (size_t)(b * SS + key) * (KVH * HDH) + g * HDH;
            v16bf b0 = make_frag(kr + qb,      kr + 16 + qb);
            v16bf b1 = make_frag(kr + 32 + qb, kr + 48 + qb);
            st[t] = wmma_bf16(q0, b0, st[t]);
            st[t] = wmma_bf16(q1, b1, st[t]);
        }
        // ---- scale + causal mask (only chunks overlapping the diagonal) ----
        const bool domask = (kb32 + 31 > qblk * 16);
#pragma unroll
        for (int t = 0; t < 2; ++t)
#pragma unroll
            for (int j = 0; j < 8; ++j) {
                float sv = st[t][j] * scale;
                if (domask) {
                    const int kc = kb32 + t * 16 + r15;
                    if (kc > qrow0 + j) sv = -1e30f;
                }
                st[t][j] = sv;
            }
        // ---- online softmax (fp32 stats, rows reduced across 16 lanes) ----
        float alph[8];
#pragma unroll
        for (int j = 0; j < 8; ++j) {
            const float rm   = redmax16(fmaxf(st[0][j], st[1][j]));
            const float newm = fmaxf(mrow[j], rm);
            alph[j] = __expf(mrow[j] - newm);
            const float p0 = __expf(st[0][j] - newm);
            const float p1 = __expf(st[1][j] - newm);
            st[0][j] = p0;
            st[1][j] = p1;
            lrow[j] = lrow[j] * alph[j] + redsum16(p0 + p1);
            mrow[j] = newm;
        }
#pragma unroll
        for (int t = 0; t < 4; ++t)
#pragma unroll
            for (int j = 0; j < 8; ++j) oacc[t][j] *= alph[j];

        // ---- relayout P: D layout -> A layout via per-wave LDS bounce ----
#pragma unroll
        for (int t = 0; t < 2; ++t)
#pragma unroll
            for (int j = 0; j < 8; ++j)
                Plds[wave][(lane >> 4) * 8 + j][t * 16 + r15] = f2bf(st[t][j]);
        asm volatile("" ::: "memory");  // same-wave LDS is in-order; block reordering
        const unsigned short* pr = &Plds[wave][r15][0];
        const v16bf pf = make_frag(pr + kb2, pr + kb2 + 8);

        // ---- P @ V: 4 output tiles across HD=64 ----
        const unsigned short* vr =
            vv + (size_t)(b * SS + kb32 + kb2) * (KVH * HDH) + g * HDH;
#pragma unroll
        for (int t = 0; t < 4; ++t) {
            v16bf vf;
#pragma unroll
            for (int i = 0; i < 16; ++i)
                vf[i] = __builtin_bit_cast(
                    __bf16, vr[(size_t)i * (KVH * HDH) + t * 16 + r15]);
            oacc[t] = wmma_bf16(pf, vf, oacc[t]);
        }
    }

    // ---- finalize: divide by row sums, store bf16 (b, s, h*HD + hd) ----
#pragma unroll
    for (int t = 0; t < 4; ++t)
#pragma unroll
        for (int j = 0; j < 8; ++j) {
            const int srow = qrow0 + j;
            const int col  = h * HDH + t * 16 + r15;
            const float v  = oacc[t][j] / lrow[j];
            ov[(size_t)(b * SS + srow) * (HH * HDH) + col] = f2bf(v);
        }
}

// ---------------------------------------------------------------------------
// Launch: bf16 pre-pass, QKV GEMMs, RoPE, attention, out-projection GEMM.
// ---------------------------------------------------------------------------
extern "C" void kernel_launch(void* const* d_in, const int* in_sizes, int n_in,
                              void* d_out, int out_size, void* d_ws, size_t ws_size,
                              hipStream_t stream) {
    const float* x  = (const float*)d_in[0];
    const float* fc = (const float*)d_in[1];
    const float* fs = (const float*)d_in[2];
    const float* wq = (const float*)d_in[3];
    const float* wk = (const float*)d_in[4];
    const float* wv = (const float*)d_in[5];
    const float* wo = (const float*)d_in[6];
    float* out = (float*)d_out;

    const size_t M = (size_t)BB * SS;              // 4096 token rows
    char* ws = (char*)d_ws;
    // bf16 staging areas (bytes)
    const size_t XB  = M * DD * 2;                 // 16 MB (x, reused for oattn)
    const size_t WQB = (size_t)(HH * HDH) * DD * 2;   // 8 MB
    const size_t WKB = (size_t)(KVH * HDH) * DD * 2;  // 2 MB
    const size_t QB  = M * (HH * HDH) * 2;         // 16 MB
    const size_t KB  = M * (KVH * HDH) * 2;        // 4 MB

    unsigned short* xbf  = (unsigned short*)(ws);
    unsigned short* oatt = xbf;  // x is dead after QKV GEMMs; reuse its region
    unsigned short* wqbf = (unsigned short*)(ws + XB);
    unsigned short* wkbf = (unsigned short*)(ws + XB + WQB);
    unsigned short* wvbf = (unsigned short*)(ws + XB + WQB + WKB);
    unsigned short* wobf = (unsigned short*)(ws + XB + WQB + 2 * WKB);
    unsigned short* qtmp = (unsigned short*)(ws + XB + 2 * WQB + 2 * WKB);
    unsigned short* ktmp = (unsigned short*)(ws + XB + 2 * WQB + 2 * WKB + QB);
    unsigned short* vtmp = (unsigned short*)(ws + XB + 2 * WQB + 2 * WKB + QB + KB);

    dim3 blk(256);
    auto cvt = [&](const float* s, unsigned short* d, size_t n) {
        const int n8 = (int)(n / 8);
        cvt_bf16_kernel<<<(n8 + 255) / 256, blk, 0, stream>>>(s, d, n8);
    };
    // One streaming pass: everything the GEMMs touch becomes bf16.
    cvt(x,  xbf,  M * DD);
    cvt(wq, wqbf, (size_t)(HH * HDH) * DD);
    cvt(wk, wkbf, (size_t)(KVH * HDH) * DD);
    cvt(wv, wvbf, (size_t)(KVH * HDH) * DD);
    cvt(wo, wobf, (size_t)DD * (HH * HDH));

    // QKV projections: C = x @ W^T (async-DMA double-buffered WMMA GEMM)
    gemm_bf16_kernel<unsigned short>
        <<<dim3((HH * HDH) / 64, M / 128), blk, 0, stream>>>(xbf, wqbf, qtmp,
                                                             (int)M, HH * HDH, DD);
    gemm_bf16_kernel<unsigned short>
        <<<dim3((KVH * HDH) / 64, M / 128), blk, 0, stream>>>(xbf, wkbf, ktmp,
                                                              (int)M, KVH * HDH, DD);
    gemm_bf16_kernel<unsigned short>
        <<<dim3((KVH * HDH) / 64, M / 128), blk, 0, stream>>>(xbf, wvbf, vtmp,
                                                              (int)M, KVH * HDH, DD);
    // RoPE on q and k in place
    const int rope_total = BB * SS * (HH + KVH) * (HDH / 2);
    rope_kernel<<<(rope_total + 255) / 256, blk, 0, stream>>>(qtmp, ktmp, fc, fs);
    // Flash attention: B*H*(S/16) waves, 8 waves per block
    attn_kernel<<<(BB * HH * (SS / 16)) / 8, blk, 0, stream>>>(qtmp, ktmp, vtmp, oatt);
    // Output projection: out = oatt @ wo^T (fp32 result)
    gemm_bf16_kernel<float>
        <<<dim3(DD / 64, M / 128), blk, 0, stream>>>(oatt, wobf, out,
                                                     (int)M, DD, HH * HDH);
}
